// Mamba2Block_43250320671443
// MI455X (gfx1250) — matrix operand order repaired and checked by reading, not compiled, not run on hardware
//
#include <hip/hip_runtime.h>
#include <hip/hip_bf16.h>

// ---------------- problem constants ----------------
#define DIMC     192
#define DINNER   768
#define HD       96
#define NH       8
#define DSTATE   64
#define DCONV    4
#define CONVDIM  896           // DINNER + 2*DSTATE
#define DPROJ    1672          // 2*DINNER + 2*DSTATE + NH
#define DPROJPAD 1728          // ceil(1672/64)*64, zero-padded B rows for GEMM1
#define NSEQV    4
#define SEQL     2048
#define LTOT     (NSEQV*SEQL)  // 8192
#define L2TOT    (2*LTOT)      // 16384 (fwd rows [0,L), bwd rows [L,2L))
#define KCHUNK   64            // k-halves staged into LDS per chunk

typedef __attribute__((ext_vector_type(16))) __bf16 v16bf;
typedef __attribute__((ext_vector_type(8)))  float  v8f;
typedef __attribute__((ext_vector_type(4)))  int    v4i;

union ABFrag { uint4 q[2]; v16bf v; };

#define AS_GLOBAL __attribute__((address_space(1)))
#define AS_LDS    __attribute__((address_space(3)))

__device__ __forceinline__ void async_load_b128_to_lds(const unsigned short* gsrc,
                                                       unsigned short* ldst) {
#if __has_builtin(__builtin_amdgcn_global_load_async_to_lds_b128)
    __builtin_amdgcn_global_load_async_to_lds_b128(
        (AS_GLOBAL v4i*)(void*)gsrc, (AS_LDS v4i*)(void*)ldst, 0, 0);
#else
    unsigned int loff = (unsigned int)(size_t)(AS_LDS char*)(void*)ldst;
    asm volatile("global_load_async_to_lds_b128 %0, %1, off"
                 :: "v"(loff), "v"(gsrc) : "memory");
#endif
}

__device__ __forceinline__ void wait_asynccnt_0() {
#if __has_builtin(__builtin_amdgcn_s_wait_asynccnt)
    __builtin_amdgcn_s_wait_asynccnt(0);
#else
    asm volatile("s_wait_asynccnt 0x0" ::: "memory");
#endif
}

__device__ __forceinline__ void wait_asynccnt_2() {
#if __has_builtin(__builtin_amdgcn_s_wait_asynccnt)
    __builtin_amdgcn_s_wait_asynccnt(2);
#else
    asm volatile("s_wait_asynccnt 0x2" ::: "memory");
#endif
}

__device__ __forceinline__ unsigned short f32_to_bf16(float f) {
    unsigned int u = __float_as_uint(f);
    unsigned int r = (u + 0x7FFFu + ((u >> 16) & 1u)) >> 16;   // RNE
    return (unsigned short)r;
}

__device__ __forceinline__ float sigmoidf_(float v) {
    return 1.0f / (1.0f + __expf(-v));
}

__device__ __forceinline__ ABFrag load_a_frag(const unsigned short* arow, int k0, int hi) {
    ABFrag a;
    a.q[0] = *(const uint4*)(arow + k0 + hi * 16);
    a.q[1] = *(const uint4*)(arow + k0 + hi * 16 + 8);
    return a;
}

__device__ __forceinline__ ABFrag load_b_frag_lds(const unsigned short* bcol, int kk, int hi) {
    ABFrag b;
    const uint4* p = (const uint4*)(bcol + kk + hi * 16);
    b.q[0] = p[0]; b.q[1] = p[1];
    return b;
}

// ---------------- K0: fp32 -> bf16 weight convert with zero row padding ----------------
__global__ void k_cvt_pad(const float* __restrict__ src,
                          unsigned short* __restrict__ dst,
                          int rows, int K, int rowsPad) {
    int i = blockIdx.x * blockDim.x + threadIdx.x;
    if (i >= rowsPad * K) return;
    int r = i / K;
    dst[i] = (r < rows) ? f32_to_bf16(src[i]) : (unsigned short)0;
}

// ---------------- K1: layernorm, one wave32 per row ----------------
__global__ void k_layernorm(const float* __restrict__ x,
                            const float* __restrict__ g,
                            const float* __restrict__ b,
                            float* __restrict__ xn) {
    int t = blockIdx.x;
    int lane = threadIdx.x;
    const float* xr = x + (size_t)t * DIMC;
    float v[6];
    float s = 0.f;
#pragma unroll
    for (int i = 0; i < 6; i++) { v[i] = xr[lane + 32 * i]; s += v[i]; }
#pragma unroll
    for (int off = 16; off; off >>= 1) s += __shfl_xor(s, off, 32);
    float mu = s * (1.0f / DIMC);
    float q = 0.f;
#pragma unroll
    for (int i = 0; i < 6; i++) { float d = v[i] - mu; q += d * d; }
#pragma unroll
    for (int off = 16; off; off >>= 1) q += __shfl_xor(q, off, 32);
    float rstd = rsqrtf(q * (1.0f / DIMC) + 1e-5f);
    float* o = xn + (size_t)t * DIMC;
#pragma unroll
    for (int i = 0; i < 6; i++) {
        int c = lane + 32 * i;
        o[c] = (v[i] - mu) * rstd * g[c] + b[c];
    }
}

// ---------------- K2: build doubled bf16 activation (fwd + reversed) ----------------
__global__ void k_build_a2(const float* __restrict__ xn,
                           const int* __restrict__ seq_idx,
                           const int* __restrict__ cu,
                           unsigned short* __restrict__ a2) {
    int r = blockIdx.x;           // [0, 2L)
    int c = threadIdx.x;          // [0, 192)
    int src;
    if (r < LTOT) {
        src = r;
    } else {
        int u = r - LTOT;
        int s = seq_idx[u];
        src = cu[s] + cu[s + 1] - 1 - u;   // reversed-segment gather
    }
    a2[(size_t)r * DIMC + c] = f32_to_bf16(xn[(size_t)src * DIMC + c]);
}

// ---------------- WMMA bf16 GEMM with async-LDS B staging ----------------
// C[M,N] = A[M,K] * B[Npad,K]^T. Block = 8 waves; wave w owns M-tile
// blockIdx.x*8+w; all waves share the 64-column B strip blockIdx.y, which is
// double-buffer streamed into LDS with GLOBAL_LOAD_ASYNC_TO_LDS_B128
// (ASYNCcnt / s_wait_asynccnt). K multiple of KCHUNK(64); Npad multiple of 64.
__global__ void k_gemm_lds(const unsigned short* __restrict__ A,
                           const unsigned short* __restrict__ B,
                           float* __restrict__ C, int N, int K) {
    __shared__ unsigned short sB[2][64 * KCHUNK];     // 2 x 8KB

    int tid  = threadIdx.x;                 // 0..255
    int wave = tid >> 5;
    int lane = tid & 31;
    int hi = lane >> 4, m = lane & 15;
    int mt  = blockIdx.x * 8 + wave;
    int nt0 = blockIdx.y * 4;               // strip of 4 n-tiles (64 cols)
    int brow0 = blockIdx.y * 64;

    const unsigned short* arow = A + (size_t)(mt * 16 + m) * K;
    const int nch = K / KCHUNK;

    // stage chunk c of the B strip into sB[c&1]: 64 rows x 64 halves
    auto stage = [&](int c) {
        const unsigned short* gbase = B + (size_t)brow0 * K + c * KCHUNK;
        unsigned short* lbase = &sB[c & 1][0];
#pragma unroll
        for (int i = 0; i < 2; i++) {
            int linear = tid + i * 256;          // 0..511
            int row = linear >> 3;               // 0..63
            int q   = linear & 7;                // b128 slot within row
            async_load_b128_to_lds(gbase + (size_t)row * K + q * 8,
                                   lbase + row * KCHUNK + q * 8);
        }
    };

    v8f acc[4] = {v8f{}, v8f{}, v8f{}, v8f{}};
    stage(0);
    for (int c = 0; c < nch; c++) {
        if (c + 1 < nch) { stage(c + 1); wait_asynccnt_2(); }
        else             { wait_asynccnt_0(); }
        __syncthreads();
        const unsigned short* bb = &sB[c & 1][0];
        int kbase = c * KCHUNK;
#pragma unroll
        for (int ks = 0; ks < KCHUNK / 32; ks++) {
            ABFrag a = load_a_frag(arow, kbase + ks * 32, hi);
#pragma unroll
            for (int j = 0; j < 4; j++) {
                ABFrag b = load_b_frag_lds(bb + (j * 16 + m) * KCHUNK, ks * 32, hi);
                acc[j] = __builtin_amdgcn_wmma_f32_16x16x32_bf16(
                    false, a.v, false, b.v, (short)0, acc[j], false, false);
            }
        }
        __syncthreads();
    }
#pragma unroll
    for (int j = 0; j < 4; j++) {
        int col = (nt0 + j) * 16 + m;
        if (col < N) {
#pragma unroll
            for (int i = 0; i < 8; i++) {
                int rr = mt * 16 + hi * 8 + i;
                C[(size_t)rr * N + col] = acc[j][i];
            }
        }
    }
}

// ---------------- K4: depthwise causal conv4 + SiLU over doubled stream ----------------
__global__ void k_conv(const float* __restrict__ zx,
                       const float* __restrict__ cw,
                       const float* __restrict__ cb,
                       const int* __restrict__ seq_idx,
                       float* __restrict__ xbc) {
    long long idx = (long long)blockIdx.x * blockDim.x + threadIdx.x;
    if (idx >= (long long)L2TOT * CONVDIM) return;
    int r = (int)(idx / CONVDIM);
    int c = (int)(idx % CONVDIM);
    int sid = (r < LTOT) ? seq_idx[r] : (NSEQV + seq_idx[r - LTOT]);
    float acc = cb[c];
#pragma unroll
    for (int k = 0; k < DCONV; k++) {
        int rr = r - (DCONV - 1 - k);
        if (rr >= 0) {
            int sid2 = (rr < LTOT) ? seq_idx[rr] : (NSEQV + seq_idx[rr - LTOT]);
            if (sid2 == sid)
                acc += cw[c * DCONV + k] * zx[(size_t)rr * DPROJ + DINNER + c];
        }
    }
    xbc[idx] = acc * sigmoidf_(acc);   // SiLU
}

// ---------------- K4b: dt = softplus(raw + bias); da = exp(dt * -exp(A_log)) ----------------
__global__ void k_dt(const float* __restrict__ zx,
                     const float* __restrict__ dtb,
                     const float* __restrict__ A_log,
                     float* __restrict__ dt, float* __restrict__ da) {
    int idx = blockIdx.x * blockDim.x + threadIdx.x;   // 2L * NH
    if (idx >= L2TOT * NH) return;
    int r = idx >> 3, h = idx & 7;
    float raw = zx[(size_t)r * DPROJ + (DPROJ - NH) + h] + dtb[h];
    float sp = (raw > 20.f) ? raw : log1pf(__expf(raw));
    dt[idx] = sp;
    da[idx] = __expf(sp * (-__expf(A_log[h])));
}

// ---------------- K5: SSM scan, one wave32 per (seq, head, p) ----------------
__global__ void k_scan(const float* __restrict__ xbc,
                       const float* __restrict__ dt,
                       const float* __restrict__ da,
                       const float* __restrict__ Dp,
                       float* __restrict__ y) {
    int bx = blockIdx.x;                       // 8 * 8 * 96
    int seq = bx / (NH * HD);
    int rem = bx % (NH * HD);
    int h = rem / HD;
    int p = rem % HD;
    int lane = threadIdx.x;
    int n0 = lane * 2;                         // 2 state entries per lane
    float h0 = 0.f, h1 = 0.f;
    float Dv = Dp[h];
    size_t rbase = (size_t)seq * SEQL;
    for (int t = 0; t < SEQL; t++) {
        size_t r = rbase + t;
        const float* xrow = xbc + r * CONVDIM;
        float2 Bv = *(const float2*)(xrow + DINNER + n0);
        float2 Cv = *(const float2*)(xrow + DINNER + DSTATE + n0);
        float dav = da[r * NH + h];
        float dtv = dt[r * NH + h];
        float xv  = xrow[h * HD + p];
        float coef = dtv * xv;
        h0 = h0 * dav + coef * Bv.x;
        h1 = h1 * dav + coef * Bv.y;
        float part = h0 * Cv.x + h1 * Cv.y;
#pragma unroll
        for (int off = 16; off; off >>= 1) part += __shfl_xor(part, off, 32);
        if (lane == 0) y[r * DINNER + h * HD + p] = part + Dv * xv;
    }
}

// ---------------- K6a: gated RMS norm -> bf16 ----------------
__global__ void k_gate_rms(const float* __restrict__ y,
                           const float* __restrict__ zx,
                           const float* __restrict__ nw,
                           unsigned short* __restrict__ yz) {
    int r = blockIdx.x;
    int tid = threadIdx.x;                 // 256 threads, 3 elems each
    __shared__ float red[8];
    float g[3];
    float ss = 0.f;
#pragma unroll
    for (int e = 0; e < 3; e++) {
        int k = tid + 256 * e;
        float zv = zx[(size_t)r * DPROJ + k];
        float gv = y[(size_t)r * DINNER + k] * (zv * sigmoidf_(zv));
        g[e] = gv; ss += gv * gv;
    }
#pragma unroll
    for (int off = 16; off; off >>= 1) ss += __shfl_xor(ss, off, 32);
    if ((tid & 31) == 0) red[tid >> 5] = ss;
    __syncthreads();
    float total = 0.f;
#pragma unroll
    for (int w = 0; w < 8; w++) total += red[w];
    float rstd = rsqrtf(total * (1.0f / DINNER) + 1e-5f);
#pragma unroll
    for (int e = 0; e < 3; e++) {
        int k = tid + 256 * e;
        yz[(size_t)r * DINNER + k] = f32_to_bf16(g[e] * rstd * nw[k]);
    }
}

// ---------------- K7a: cat = [fwd(t), bwd(rev(t))] -> bf16 ----------------
__global__ void k_cat(const float* __restrict__ yp,
                      const int* __restrict__ seq_idx,
                      const int* __restrict__ cu,
                      unsigned short* __restrict__ cat) {
    int t = blockIdx.x;            // [0, L)
    int c = threadIdx.x;           // [0, 384)
    float v;
    if (c < DIMC) {
        v = yp[(size_t)t * DIMC + c];
    } else {
        int s = seq_idx[t];
        int rt = cu[s] + cu[s + 1] - 1 - t;
        v = yp[(size_t)(LTOT + rt) * DIMC + (c - DIMC)];
    }
    cat[(size_t)t * (2 * DIMC) + c] = f32_to_bf16(v);
}

// ---------------- K7b: final GEMM (16x64 strips) + bias + residual ----------------
// M=L, N=192 (3 strips of 64), K=384. Writes both tuple copies of d_out.
__global__ void k_gemm_final(const unsigned short* __restrict__ A,   // L x 384
                             const unsigned short* __restrict__ B,   // 192 x 384
                             const float* __restrict__ x,
                             const float* __restrict__ bias,
                             float* __restrict__ out) {
    const int K = 2 * DIMC;       // 384
    int mt  = blockIdx.x;
    int nt0 = blockIdx.y * 4;
    int lane = threadIdx.x & 31;
    int hi = lane >> 4, m = lane & 15;
    const unsigned short* arow = A + (size_t)(mt * 16 + m) * K;
    const unsigned short* bcol = B + (size_t)(nt0 * 16 + m) * K;
    const size_t bstride = (size_t)16 * K;
    v8f acc[4] = {v8f{}, v8f{}, v8f{}, v8f{}};
    for (int k0 = 0; k0 < K; k0 += 32) {
        ABFrag a = load_a_frag(arow, k0, hi);
#pragma unroll
        for (int j = 0; j < 4; j++) {
            ABFrag b;
            const uint4* p = (const uint4*)(bcol + (size_t)j * bstride + k0 + hi * 16);
            b.q[0] = p[0]; b.q[1] = p[1];
            acc[j] = __builtin_amdgcn_wmma_f32_16x16x32_bf16(
                false, a.v, false, b.v, (short)0, acc[j], false, false);
        }
    }
#pragma unroll
    for (int j = 0; j < 4; j++) {
        int col = (nt0 + j) * 16 + m;
        float bv = bias[col];
#pragma unroll
        for (int i = 0; i < 8; i++) {
            int rr = mt * 16 + hi * 8 + i;
            float v = acc[j][i] + bv + x[(size_t)rr * DIMC + col];
            out[(size_t)rr * DIMC + col] = v;                    // tuple elem 0
            out[(size_t)(LTOT + rr) * DIMC + col] = v;           // tuple elem 1 (same)
        }
    }
}

// ---------------- host launcher ----------------
extern "C" void kernel_launch(void* const* d_in, const int* in_sizes, int n_in,
                              void* d_out, int out_size, void* d_ws, size_t ws_size,
                              hipStream_t stream) {
    const float* x          = (const float*)d_in[0];
    const float* ln_gamma   = (const float*)d_in[1];
    const float* ln_beta    = (const float*)d_in[2];
    const float* in_proj_w  = (const float*)d_in[3];   // 1672 x 192
    const float* conv_w     = (const float*)d_in[4];   // 896 x 4
    const float* conv_b     = (const float*)d_in[5];
    const float* A_log      = (const float*)d_in[6];
    const float* dt_bias    = (const float*)d_in[7];
    const float* Dp         = (const float*)d_in[8];
    const float* norm_w     = (const float*)d_in[9];
    const float* out_proj_w = (const float*)d_in[10];  // 192 x 768
    const float* out_prj2_w = (const float*)d_in[11];  // 192 x 384
    const float* out_prj2_b = (const float*)d_in[12];
    const int*   seq_idx    = (const int*)d_in[13];
    const int*   cu         = (const int*)d_in[14];
    float* out = (float*)d_out;

    // ---- carve workspace (all offsets 256B aligned) ----
    char* W = (char*)d_ws;
    size_t off = 0;
    auto take = [&](size_t bytes) {
        size_t o = off;
        off = (off + bytes + 255) & ~(size_t)255;
        return o;
    };
    float*          xn    = (float*)         (W + take((size_t)LTOT    * DIMC   * 4));
    unsigned short* a2    = (unsigned short*)(W + take((size_t)L2TOT   * DIMC   * 2));
    unsigned short* win   = (unsigned short*)(W + take((size_t)DPROJPAD* DIMC   * 2));
    unsigned short* wout  = (unsigned short*)(W + take((size_t)DIMC    * DINNER * 2));
    unsigned short* wp2   = (unsigned short*)(W + take((size_t)DIMC    * 2*DIMC * 2));
    float*          zx    = (float*)         (W + take((size_t)L2TOT   * DPROJ  * 4));
    float*          xbc   = (float*)         (W + take((size_t)L2TOT   * CONVDIM* 4));
    float*          dtb   = (float*)         (W + take((size_t)L2TOT   * NH     * 4));
    float*          dab   = (float*)         (W + take((size_t)L2TOT   * NH     * 4));
    float*          ybuf  = (float*)         (W + take((size_t)L2TOT   * DINNER * 4));
    unsigned short* yz    = (unsigned short*)(W + take((size_t)L2TOT   * DINNER * 2));
    float*          yproj = (float*)         (W + take((size_t)L2TOT   * DIMC   * 4));
    unsigned short* cat   = (unsigned short*)(W + take((size_t)LTOT    * 2*DIMC * 2));
    (void)ws_size; (void)in_sizes; (void)n_in; (void)out_size;

    // 0) weights -> bf16 (in_proj zero-padded to 1728 rows for branch-free GEMM)
    {
        int n1 = DPROJPAD * DIMC, n2 = DIMC * DINNER, n3 = DIMC * 2 * DIMC;
        k_cvt_pad<<<(n1 + 255) / 256, 256, 0, stream>>>(in_proj_w,  win,  DPROJ, DIMC, DPROJPAD);
        k_cvt_pad<<<(n2 + 255) / 256, 256, 0, stream>>>(out_proj_w, wout, DIMC, DINNER, DIMC);
        k_cvt_pad<<<(n3 + 255) / 256, 256, 0, stream>>>(out_prj2_w, wp2,  DIMC, 2 * DIMC, DIMC);
    }
    // 1) layernorm
    k_layernorm<<<LTOT, 32, 0, stream>>>(x, ln_gamma, ln_beta, xn);
    // 2) doubled bf16 activations (fwd rows + reversed-segment rows)
    k_build_a2<<<L2TOT, DIMC, 0, stream>>>(xn, seq_idx, cu, a2);
    // 3) in_proj GEMM: (2L x 192) x (1728pad x 192)^T -> (2L x 1672); async-LDS B
    k_gemm_lds<<<dim3(L2TOT / 128, DPROJPAD / 64), 256, 0, stream>>>(a2, win, zx, DPROJ, DIMC);
    // 4) conv + SiLU
    {
        long long n = (long long)L2TOT * CONVDIM;
        k_conv<<<(unsigned)((n + 255) / 256), 256, 0, stream>>>(zx, conv_w, conv_b, seq_idx, xbc);
    }
    // 5) dt / da
    k_dt<<<(L2TOT * NH + 255) / 256, 256, 0, stream>>>(zx, dt_bias, A_log, dtb, dab);
    // 6) SSM scan (8 sequences x 8 heads x 96 headdim wave32s)
    k_scan<<<2 * NSEQV * NH * HD, 32, 0, stream>>>(xbc, dtb, dab, Dp, ybuf);
    // 7) gated RMS norm -> bf16
    k_gate_rms<<<L2TOT, 256, 0, stream>>>(ybuf, zx, norm_w, yz);
    // 8) out_proj GEMM: (2L x 768) x (192 x 768)^T -> (2L x 192); async-LDS B
    k_gemm_lds<<<dim3(L2TOT / 128, DIMC / 64), 256, 0, stream>>>(yz, wout, yproj, DIMC, DINNER);
    // 9) cat fwd + reverse-back bwd
    k_cat<<<LTOT, 2 * DIMC, 0, stream>>>(yproj, seq_idx, cu, cat);
    // 10) final GEMM + bias + residual -> both tuple copies of d_out
    k_gemm_final<<<dim3(LTOT / 16, DIMC / 64), 32, 0, stream>>>(cat, wp2, x, out_prj2_b, out);
}
